// EncoderPp_24472723653374
// MI455X (gfx1250) — compile-verified, matrix operand order checked by reading, not run on hardware
//
#include <hip/hip_runtime.h>
#include <cstdint>
#include <cstddef>

// ---------------------------------------------------------------------------
// MI455X (gfx1250, wave32) implementation of the PointNet++-style encoder.
// Dense math: v_wmma_f32_16x16x32_bf16 (f32 accumulate).
// Data movement: async global->LDS copies (ASYNCcnt) for gathers / A panels.
// ---------------------------------------------------------------------------

typedef __bf16 bf16_t;
typedef bf16_t v16bf __attribute__((ext_vector_type(16)));
typedef float  v8f   __attribute__((ext_vector_type(8)));
typedef unsigned short u16;

static constexpr int BB   = 8;
static constexpr int NN   = 4096;
static constexpr int M1   = 1024;
static constexpr int M2   = 256;
static constexpr int KNBR = 128;

#if __has_builtin(__builtin_amdgcn_global_load_async_to_lds_b128)
#define HAVE_ASYNC_LDS 1
#else
#define HAVE_ASYNC_LDS 0
#endif

// ---- helpers ---------------------------------------------------------------

__device__ __forceinline__ u16 f2bf(float f) {
  // round-to-nearest-even truncation to bf16 bits
  uint32_t x = __float_as_uint(f);
  uint32_t r = (x + 0x7FFFu + ((x >> 16) & 1u)) >> 16;
  return (u16)r;
}

// 16-byte copy global -> LDS. On gfx1250 this lowers to
// global_load_async_to_lds_b128 (no VGPR round trip, tracked by ASYNCcnt).
// Builtin signature (per clang-22 diagnostic): (v4i addrspace(1)*,
// v4i addrspace(3)*, imm offset, imm cpol).
__device__ __forceinline__ void async_copy_b128(const u16* g, u16* l) {
#if HAVE_ASYNC_LDS
  typedef int v4i_t __attribute__((vector_size(16)));
  typedef __attribute__((address_space(1))) v4i_t* gp_t;
  typedef __attribute__((address_space(3))) v4i_t* lp_t;
  __builtin_amdgcn_global_load_async_to_lds_b128((gp_t)g, (lp_t)l, 0, 0);
#else
  *(uint4*)l = *(const uint4*)g;
#endif
}

__device__ __forceinline__ void wait_async() {
#if HAVE_ASYNC_LDS
#if __has_builtin(__builtin_amdgcn_s_wait_asynccnt)
  __builtin_amdgcn_s_wait_asynccnt(0);
#else
  asm volatile("s_wait_asynccnt 0x0" ::: "memory");
#endif
#endif
}

union FragU { v16bf v; uint4 q[2]; };

// A-fragment (16x32 bf16) from a row-major [rows x ldk] bf16 buffer (LDS or
// global). ISA layout: lanes 0-15 -> kb=0, lanes 16-31 -> kb=8; halves 0..7 are
// K=k0+kb..+7, halves 8..15 are K=k0+16+kb..+7.  Two aligned 16B loads/lane.
__device__ __forceinline__ v16bf load_rm_frag(const u16* base, int ldk,
                                              int row0, int k0, int lane) {
  int r  = row0 + (lane & 15);
  int kb = (lane & 16) ? 8 : 0;
  const u16* p = base + (size_t)r * ldk + k0 + kb;
  FragU u;
  u.q[0] = *(const uint4*)(p);
  u.q[1] = *(const uint4*)(p + 16);
  return u.v;
}

// B-fragment from fragment-packed weight buffer: [frag][lane][16] halves.
__device__ __forceinline__ v16bf load_packed_frag(const u16* frags,
                                                  int fragIdx, int lane) {
  const u16* p = frags + ((size_t)fragIdx * 32 + lane) * 16;
  FragU u;
  u.q[0] = ((const uint4*)p)[0];
  u.q[1] = ((const uint4*)p)[1];
  return u.v;
}

__device__ __forceinline__ v8f wmma_bf16(v16bf a, v16bf b, v8f c) {
  return __builtin_amdgcn_wmma_f32_16x16x32_bf16(false, a, false, b,
                                                 (short)0, c, false, false);
}

// ---- weight packing: f32 (K x N) -> bf16 WMMA B fragments ------------------
__global__ __launch_bounds__(256) void pack_wmma_b_kernel(
    const float* __restrict__ W, u16* __restrict__ dst,
    int Kdim, int Kpad, int Nout) {
  int total = (Kpad / 32) * (Nout / 16) * 32 * 16;
  for (int i = blockIdx.x * blockDim.x + threadIdx.x; i < total;
       i += gridDim.x * blockDim.x) {
    int h    = i & 15;
    int lane = (i >> 4) & 31;
    int frag = i >> 9;
    int ntil = frag % (Nout / 16);
    int ks   = frag / (Nout / 16);
    int n    = ntil * 16 + (lane & 15);
    int kb   = (lane & 16) ? 8 : 0;
    int k    = ks * 32 + ((h < 8) ? (kb + h) : (16 + kb + (h - 8)));
    float v  = (k < Kdim) ? W[(size_t)k * Nout + n] : 0.0f;
    dst[i] = f2bf(v);
  }
}

// ---- local MLP: pos[B,N,2] -> local[B,N,64] (tanh both layers) -------------
// Also builds gin bf16 [B,N,96]: cols 0..63 local, 64..66 zones, 67..95 zero.
__global__ __launch_bounds__(256) void local_mlp_kernel(
    const float* __restrict__ pos, const float* __restrict__ zones,
    const float* __restrict__ w1, const float* __restrict__ b1,
    const float* __restrict__ w2, const float* __restrict__ b2,
    float* __restrict__ outLocal, u16* __restrict__ gin) {
  __shared__ float sW1[2 * 64];
  __shared__ float sB1[64];
  __shared__ float sW2[64 * 64];
  __shared__ float sB2[64];
  int t = threadIdx.x;
  for (int i = t; i < 128; i += 256) sW1[i] = w1[i];
  for (int i = t; i < 64; i += 256) { sB1[i] = b1[i]; sB2[i] = b2[i]; }
  for (int i = t; i < 4096; i += 256) sW2[i] = w2[i];
  __syncthreads();

  int p = blockIdx.x * 256 + t;           // point index in [0, B*N)
  float px = pos[(size_t)p * 2 + 0];
  float py = pos[(size_t)p * 2 + 1];
  float h1[64];
#pragma unroll
  for (int j = 0; j < 64; ++j)
    h1[j] = tanhf(px * sW1[j] + py * sW1[64 + j] + sB1[j]);
  for (int j = 0; j < 64; ++j) {
    float a = sB2[j];
#pragma unroll 8
    for (int i = 0; i < 64; ++i) a += h1[i] * sW2[i * 64 + j];
    float h = tanhf(a);
    outLocal[(size_t)p * 64 + j] = h;
    gin[(size_t)p * 96 + j] = f2bf(h);
  }
#pragma unroll
  for (int k = 0; k < 3; ++k)
    gin[(size_t)p * 96 + 64 + k] = f2bf(zones[(size_t)p * 3 + k]);
#pragma unroll
  for (int c = 67; c < 96; ++c) gin[(size_t)p * 96 + c] = 0;
}

// ---- farthest point sampling (deterministic, start at 0) -------------------
__global__ __launch_bounds__(1024) void fps_kernel(
    const float* __restrict__ pos, int n, int m,
    int* __restrict__ idx, float* __restrict__ q) {
  int b = blockIdx.x, t = threadIdx.x;
  const float* P = pos + (size_t)b * n * 2;
  int pt = n >> 10;  // points per thread (<= 4)
  float px[4], py[4], mind[4];
  for (int j = 0; j < pt; ++j) {
    int i = t + (j << 10);
    px[j] = P[(size_t)i * 2 + 0];
    py[j] = P[(size_t)i * 2 + 1];
    mind[j] = 3.4e38f;
  }
  __shared__ float sv[1024];
  __shared__ int   si[1024];
  __shared__ float sqx, sqy;
  int last = 0;
  for (int it = 0; it < m; ++it) {
    if (t == 0) {
      idx[(size_t)b * m + it] = last;
      float lx = P[(size_t)last * 2 + 0], ly = P[(size_t)last * 2 + 1];
      q[((size_t)b * m + it) * 2 + 0] = lx;
      q[((size_t)b * m + it) * 2 + 1] = ly;
      sqx = lx; sqy = ly;
    }
    __syncthreads();
    float lx = sqx, ly = sqy;
    float bv = -1.0f; int bi = 0;
    for (int j = 0; j < pt; ++j) {
      float dx = px[j] - lx, dy = py[j] - ly;
      float d = dx * dx + dy * dy;
      if (d < mind[j]) mind[j] = d;
      if (mind[j] > bv) { bv = mind[j]; bi = t + (j << 10); }
    }
    sv[t] = bv; si[t] = bi;
    __syncthreads();
    for (int o = 512; o > 0; o >>= 1) {
      if (t < o && sv[t + o] > sv[t]) { sv[t] = sv[t + o]; si[t] = si[t + o]; }
      __syncthreads();
    }
    last = si[0];
    __syncthreads();
  }
}

// ---- radius neighbor search (one wave32 per center) ------------------------
__global__ __launch_bounds__(256) void nbr_kernel(
    const float* __restrict__ srcpos, const float* __restrict__ q,
    int n, int m, float r2, int* __restrict__ nbrs, int* __restrict__ cnts) {
  int wave = threadIdx.x >> 5, lane = threadIdx.x & 31;
  int c = blockIdx.x * 8 + wave;
  int b = c / m;
  const float* P = srcpos + (size_t)b * n * 2;
  float qx = q[(size_t)c * 2 + 0], qy = q[(size_t)c * 2 + 1];
  int count = 0;
  for (int base = 0; base < n; base += 32) {
    int i = base + lane;
    float dx = P[(size_t)i * 2 + 0] - qx;
    float dy = P[(size_t)i * 2 + 1] - qy;
    bool ok = (dx * dx + dy * dy) <= r2;
    unsigned mask = (unsigned)__ballot(ok);   // wave32: low 32 bits
    if (ok) {
      int p = count + __popc(mask & ((1u << lane) - 1u));
      if (p < KNBR) nbrs[(size_t)c * KNBR + p] = i;
    }
    count += __popc(mask);
  }
  if (lane == 0) cnts[c] = count < KNBR ? count : KNBR;
}

// ---- set-abstraction conv: async gather -> WMMA MLP -> masked max ----------
// One 256-thread block (8 waves) per center. feat rows are bf16 with stride
// KIN, zero tail; rel written at relCol/relCol+1 during staging.
template <int KIN, int C1, int C2, int OUTSTRIDE, bool WRITEQ>
__global__ __launch_bounds__(256) void sa_conv_kernel(
    const u16* __restrict__ feat, const float* __restrict__ srcpos,
    const float* __restrict__ qarr, const int* __restrict__ nbrs,
    const int* __restrict__ cnts, int n, int m, int relCol,
    const u16* __restrict__ w1f, const float* __restrict__ b1v,
    const u16* __restrict__ w2f, const float* __restrict__ b2v,
    u16* __restrict__ outFeat) {
  extern __shared__ u16 sdyn[];
  u16* As = sdyn;                  // [128][KIN] bf16
  u16* H1 = sdyn + 128 * KIN;      // [128][C1]  bf16

  const int tid = threadIdx.x;
  const int wave = tid >> 5, lane = tid & 31;
  const int hi8 = (lane & 16) ? 8 : 0;
  const int cidx = blockIdx.x;
  const int b = cidx / m;
  const int count = cnts[cidx];
  const float qx = qarr[(size_t)cidx * 2 + 0];
  const float qy = qarr[(size_t)cidx * 2 + 1];
  const float* P = srcpos + (size_t)b * n * 2;
  const int* nb = nbrs + (size_t)cidx * KNBR;

  // Stage gathered neighbor rows into LDS. Valid rows use per-lane async
  // global->LDS b128 copies (gather path, no VGPR round trip); invalid -> 0.
  constexpr int VPR = KIN / 8;  // uint4 chunks per row
  for (int e = tid; e < 128 * VPR; e += 256) {
    int row = e / VPR;
    int off = (e - row * VPR) * 8;
    if (row < count) {
      int s = nb[row];
      async_copy_b128(feat + ((size_t)b * n + s) * KIN + off,
                      As + row * KIN + off);
    } else {
      *(uint4*)(As + row * KIN + off) = make_uint4(0u, 0u, 0u, 0u);
    }
  }
  wait_async();
  __syncthreads();
  if (tid < 128 && tid < count) {
    int s = nb[tid];
    As[tid * KIN + relCol]     = f2bf(P[(size_t)s * 2 + 0] - qx);
    As[tid * KIN + relCol + 1] = f2bf(P[(size_t)s * 2 + 1] - qy);
  }
  __syncthreads();

  // Layer 1: [128 x KIN] @ [KIN x C1] + b1, tanh -> H1 (bf16, LDS)
  constexpr int KS1 = KIN / 32;
  constexpr int NT1 = C1 / 16;
  for (int s = 0; s < NT1 / 8; ++s) {
    int nt = wave + s * 8;
    v8f z = {0, 0, 0, 0, 0, 0, 0, 0};
    v8f acc[8];
#pragma unroll
    for (int i = 0; i < 8; ++i) acc[i] = z;
    for (int ks = 0; ks < KS1; ++ks) {
      v16bf bf = load_packed_frag(w1f, ks * NT1 + nt, lane);
      if (ks + 1 < KS1)
        __builtin_prefetch(w1f + ((size_t)((ks + 1) * NT1 + nt) * 32 + lane) * 16, 0, 1);
#pragma unroll
      for (int mt = 0; mt < 8; ++mt) {
        v16bf af = load_rm_frag(As, KIN, mt * 16, ks * 32, lane);
        acc[mt] = wmma_bf16(af, bf, acc[mt]);
      }
    }
    int col = nt * 16 + (lane & 15);
    float bias = b1v[col];
#pragma unroll
    for (int mt = 0; mt < 8; ++mt)
#pragma unroll
      for (int v = 0; v < 8; ++v)
        H1[(mt * 16 + v + hi8) * C1 + col] = f2bf(tanhf(acc[mt][v] + bias));
  }
  __syncthreads();

  // Layer 2 (plain) + masked max over neighbor rows.
  constexpr int KS2 = C1 / 32;
  constexpr int NT2 = C2 / 16;
  for (int s = 0; s < NT2 / 8; ++s) {
    int nt = wave + s * 8;
    v8f z = {0, 0, 0, 0, 0, 0, 0, 0};
    v8f acc[8];
#pragma unroll
    for (int i = 0; i < 8; ++i) acc[i] = z;
    for (int ks = 0; ks < KS2; ++ks) {
      v16bf bf = load_packed_frag(w2f, ks * NT2 + nt, lane);
#pragma unroll
      for (int mt = 0; mt < 8; ++mt) {
        v16bf af = load_rm_frag(H1, C1, mt * 16, ks * 32, lane);
        acc[mt] = wmma_bf16(af, bf, acc[mt]);
      }
    }
    int col = nt * 16 + (lane & 15);
    float bias = b2v[col];
    float mv = -3.4e38f;
#pragma unroll
    for (int mt = 0; mt < 8; ++mt)
#pragma unroll
      for (int v = 0; v < 8; ++v) {
        int row = mt * 16 + v + hi8;
        if (row < count) mv = fmaxf(mv, acc[mt][v] + bias);
      }
    mv = fmaxf(mv, __shfl_xor(mv, 16));   // fold M and M+8 half-tiles
    if (lane < 16) outFeat[(size_t)cidx * OUTSTRIDE + col] = f2bf(mv);
  }
  // Zero-pad tail columns (and optionally append center coords for glob MLP).
  if (tid < OUTSTRIDE - C2) {
    int c = C2 + tid;
    u16 v = 0;
    if (WRITEQ) {
      if (c == C2) v = f2bf(qx);
      else if (c == C2 + 1) v = f2bf(qy);
    }
    outFeat[(size_t)cidx * OUTSTRIDE + c] = v;
  }
}

// ---- global MLP layer 1: [256 x 288] @ [288 x 512] + b, tanh -> bf16 -------
// 8 waves/block, one 16-col strip per wave; A K-slabs double-buffered in LDS
// via async global->LDS copies overlapping the WMMAs of the previous slab.
__global__ __launch_bounds__(256) void glob1_kernel(
    const u16* __restrict__ x2h, const u16* __restrict__ w1f,
    const float* __restrict__ b1v, u16* __restrict__ h1g) {
  __shared__ u16 slab[2][256 * 32];   // 2 x 16 KB
  int b = blockIdx.x >> 2;
  int bs = blockIdx.x & 3;
  int tid = threadIdx.x;
  int wave = tid >> 5, lane = tid & 31;
  int hi8 = (lane & 16) ? 8 : 0;
  int nt = bs * 8 + wave;  // strip 0..31
  const u16* A = x2h + (size_t)b * 256 * 288;

  auto loadSlab = [&](int ks, int buf) {
    for (int e = tid; e < 1024; e += 256) {       // 256 rows x 4 b128 chunks
      int row = e >> 2;
      int ch = (e & 3) * 8;
      async_copy_b128(A + (size_t)row * 288 + ks * 32 + ch,
                      &slab[buf][row * 32 + ch]);
    }
  };

  loadSlab(0, 0);
  wait_async();
  __syncthreads();

  v8f z = {0, 0, 0, 0, 0, 0, 0, 0};
  v8f acc[16];
#pragma unroll
  for (int i = 0; i < 16; ++i) acc[i] = z;
  for (int ks = 0; ks < 9; ++ks) {
    if (ks + 1 < 9) loadSlab(ks + 1, (ks + 1) & 1);
    v16bf bf = load_packed_frag(w1f, ks * 32 + nt, lane);
#pragma unroll
    for (int mt = 0; mt < 16; ++mt) {
      v16bf af = load_rm_frag(&slab[ks & 1][0], 32, mt * 16, 0, lane);
      acc[mt] = wmma_bf16(af, bf, acc[mt]);
    }
    wait_async();
    __syncthreads();
  }
  int col = nt * 16 + (lane & 15);
  float bias = b1v[col];
#pragma unroll
  for (int mt = 0; mt < 16; ++mt)
#pragma unroll
    for (int v = 0; v < 8; ++v) {
      int row = mt * 16 + v + hi8;
      h1g[((size_t)b * 256 + row) * 512 + col] = f2bf(tanhf(acc[mt][v] + bias));
    }
}

// ---- global MLP layer 2 + max pool over 256 rows ---------------------------
__global__ __launch_bounds__(256) void glob2_kernel(
    const u16* __restrict__ h1g, const u16* __restrict__ w2f,
    const float* __restrict__ b2v, float* __restrict__ gout) {
  __shared__ u16 slab[2][256 * 32];   // 2 x 16 KB
  int b = blockIdx.x >> 3;
  int bs = blockIdx.x & 7;
  int tid = threadIdx.x;
  int wave = tid >> 5, lane = tid & 31;
  int nt = bs * 8 + wave;  // strip 0..63
  const u16* A = h1g + (size_t)b * 256 * 512;

  auto loadSlab = [&](int ks, int buf) {
    for (int e = tid; e < 1024; e += 256) {
      int row = e >> 2;
      int ch = (e & 3) * 8;
      async_copy_b128(A + (size_t)row * 512 + ks * 32 + ch,
                      &slab[buf][row * 32 + ch]);
    }
  };

  loadSlab(0, 0);
  wait_async();
  __syncthreads();

  v8f z = {0, 0, 0, 0, 0, 0, 0, 0};
  v8f acc[16];
#pragma unroll
  for (int i = 0; i < 16; ++i) acc[i] = z;
  for (int ks = 0; ks < 16; ++ks) {
    if (ks + 1 < 16) loadSlab(ks + 1, (ks + 1) & 1);
    v16bf bf = load_packed_frag(w2f, ks * 64 + nt, lane);
    if (ks + 1 < 16)
      __builtin_prefetch(w2f + ((size_t)((ks + 1) * 64 + nt) * 32 + lane) * 16, 0, 1);
#pragma unroll
    for (int mt = 0; mt < 16; ++mt) {
      v16bf af = load_rm_frag(&slab[ks & 1][0], 32, mt * 16, 0, lane);
      acc[mt] = wmma_bf16(af, bf, acc[mt]);
    }
    wait_async();
    __syncthreads();
  }
  int col = nt * 16 + (lane & 15);
  float bias = b2v[col];
  float mv = -3.4e38f;
#pragma unroll
  for (int mt = 0; mt < 16; ++mt)
#pragma unroll
    for (int v = 0; v < 8; ++v) mv = fmaxf(mv, acc[mt][v] + bias);
  mv = fmaxf(mv, __shfl_xor(mv, 16));
  if (lane < 16) gout[(size_t)b * 1024 + col] = mv;
}

// ---------------------------------------------------------------------------

extern "C" void kernel_launch(void* const* d_in, const int* in_sizes, int n_in,
                              void* d_out, int out_size, void* d_ws, size_t ws_size,
                              hipStream_t stream) {
  (void)in_sizes; (void)n_in; (void)out_size; (void)ws_size;

  const float* pos   = (const float*)d_in[0];   // [8,4096,2]
  const float* zones = (const float*)d_in[1];   // [8,4096,3]
  const float* lw1 = (const float*)d_in[2];     // [2,64]
  const float* lb1 = (const float*)d_in[3];
  const float* lw2 = (const float*)d_in[4];     // [64,64]
  const float* lb2 = (const float*)d_in[5];
  const float* s1w1 = (const float*)d_in[6];    // [69,128]
  const float* s1b1 = (const float*)d_in[7];
  const float* s1w2 = (const float*)d_in[8];    // [128,128]
  const float* s1b2 = (const float*)d_in[9];
  const float* s2w1 = (const float*)d_in[10];   // [130,256]
  const float* s2b1 = (const float*)d_in[11];
  const float* s2w2 = (const float*)d_in[12];   // [256,256]
  const float* s2b2 = (const float*)d_in[13];
  const float* gw1 = (const float*)d_in[14];    // [258,512]
  const float* gb1 = (const float*)d_in[15];
  const float* gw2 = (const float*)d_in[16];    // [512,1024]
  const float* gb2 = (const float*)d_in[17];

  float* outLocal = (float*)d_out;                        // [8,4096,64]
  float* outGlob  = outLocal + (size_t)BB * NN * 64;      // [8,1024]

  // workspace carve-up (256B aligned)
  char* ws = (char*)d_ws;
  size_t off = 0;
  auto carve = [&](size_t bytes) -> char* {
    char* p = ws + off;
    off = (off + bytes + 255) & ~(size_t)255;
    return p;
  };
  u16*  gin   = (u16*)carve((size_t)BB * NN * 96 * 2);
  int*  idx1  = (int*)carve((size_t)BB * M1 * 4);
  float* q1   = (float*)carve((size_t)BB * M1 * 2 * 4);
  int*  nbr1  = (int*)carve((size_t)BB * M1 * KNBR * 4);
  int*  cnt1  = (int*)carve((size_t)BB * M1 * 4);
  u16*  x1h   = (u16*)carve((size_t)BB * M1 * 160 * 2);
  int*  idx2  = (int*)carve((size_t)BB * M2 * 4);
  float* q2   = (float*)carve((size_t)BB * M2 * 2 * 4);
  int*  nbr2  = (int*)carve((size_t)BB * M2 * KNBR * 4);
  int*  cnt2  = (int*)carve((size_t)BB * M2 * 4);
  u16*  x2h   = (u16*)carve((size_t)BB * M2 * 288 * 2);
  u16*  h1g   = (u16*)carve((size_t)BB * 256 * 512 * 2);
  u16*  s1w1f = (u16*)carve((size_t)3 * 8 * 512 * 2);
  u16*  s1w2f = (u16*)carve((size_t)4 * 8 * 512 * 2);
  u16*  s2w1f = (u16*)carve((size_t)5 * 16 * 512 * 2);
  u16*  s2w2f = (u16*)carve((size_t)8 * 16 * 512 * 2);
  u16*  gw1f  = (u16*)carve((size_t)9 * 32 * 512 * 2);
  u16*  gw2f  = (u16*)carve((size_t)16 * 64 * 512 * 2);

  // 1) pack weights into WMMA fragment order (bf16)
  auto packLaunch = [&](const float* W, u16* dst, int K, int Kpad, int N) {
    int total = (Kpad / 32) * (N / 16) * 512;
    pack_wmma_b_kernel<<<(total + 255) / 256, 256, 0, stream>>>(W, dst, K, Kpad, N);
  };
  packLaunch(s1w1, s1w1f, 69, 96, 128);
  packLaunch(s1w2, s1w2f, 128, 128, 128);
  packLaunch(s2w1, s2w1f, 130, 160, 256);
  packLaunch(s2w2, s2w2f, 256, 256, 256);
  packLaunch(gw1, gw1f, 258, 288, 512);
  packLaunch(gw2, gw2f, 512, 512, 1024);

  // 2) local MLP + gin staging
  local_mlp_kernel<<<(BB * NN) / 256, 256, 0, stream>>>(
      pos, zones, lw1, lb1, lw2, lb2, outLocal, gin);

  // 3) level-1 FPS + radius neighbors + PointNetConv (WMMA)
  fps_kernel<<<BB, 1024, 0, stream>>>(pos, NN, M1, idx1, q1);
  nbr_kernel<<<(BB * M1) / 8, 256, 0, stream>>>(pos, q1, NN, M1, 0.05f * 0.05f,
                                                nbr1, cnt1);
  {
    size_t shmem = (size_t)(128 * 96 + 128 * 128) * 2;  // 56 KB
    sa_conv_kernel<96, 128, 128, 160, false>
        <<<BB * M1, 256, shmem, stream>>>(gin, pos, q1, nbr1, cnt1, NN, M1, 67,
                                          s1w1f, s1b1, s1w2f, s1b2, x1h);
  }

  // 4) level-2 FPS + neighbors + conv
  fps_kernel<<<BB, 1024, 0, stream>>>(q1, M1, M2, idx2, q2);
  nbr_kernel<<<(BB * M2) / 8, 256, 0, stream>>>(q1, q2, M1, M2, 0.1f * 0.1f,
                                                nbr2, cnt2);
  {
    size_t shmem = (size_t)(128 * 160 + 128 * 256) * 2;  // 104 KB (WGP has 320 KB)
    sa_conv_kernel<160, 256, 256, 288, true>
        <<<BB * M2, 256, shmem, stream>>>(x1h, q1, q2, nbr2, cnt2, M1, M2, 128,
                                          s2w1f, s2b1, s2w2f, s2b2, x2h);
  }

  // 5) global MLP + max pool
  glob1_kernel<<<BB * 4, 256, 0, stream>>>(x2h, gw1f, gb1, h1g);
  glob2_kernel<<<BB * 8, 256, 0, stream>>>(h1g, gw2f, gb2, outGlob);
}
